// Attention_34033320854122
// MI455X (gfx1250) — compile-verified, matrix operand order checked by reading, not run on hardware
//
#include <hip/hip_runtime.h>

#define T_DIM 2048
#define D_DIM 2048
#define H_N   16
#define HD_N  128
#define RD_N  64
#define EPS_F 1e-5f

#define KB_STAGE 64          // K elements per LDS stage (2 WMMA k-steps)
#define LDSROW   72          // u16 per padded LDS row (144 B: 128 B data + 16 B pad)

typedef unsigned short u16;
typedef __attribute__((ext_vector_type(8)))  __bf16 v8bf;
typedef __attribute__((ext_vector_type(16))) __bf16 v16bf;
typedef __attribute__((ext_vector_type(8)))  float  v8f;

__device__ __forceinline__ u16 f32_to_bf16(float f) {
  unsigned u = __float_as_uint(f);
  u += 0x7FFFu + ((u >> 16) & 1u);   // round-to-nearest-even
  return (u16)(u >> 16);
}

// Load one 16x32 bf16 WMMA fragment (A or B operand, NT form).
// Per ISA 7.12.2: lanes 0-15 hold K {0..7, 16..23}, lanes 16-31 hold
// K {8..15, 24..31}. Caller passes row pointer already offset by k0 + half*8.
__device__ __forceinline__ v16bf load_frag(const u16* p) {
  v8bf lo = *reinterpret_cast<const v8bf*>(p);        // K slots 0..7
  v8bf hi = *reinterpret_cast<const v8bf*>(p + 16);   // K slots 8..15
  return __builtin_shufflevector(lo, hi, 0,1,2,3,4,5,6,7,8,9,10,11,12,13,14,15);
}

__device__ __forceinline__ v8f v8f_zero() {
  v8f z = {0.f,0.f,0.f,0.f,0.f,0.f,0.f,0.f};
  return z;
}

// Async 16-byte global -> LDS copy (CDNA5 ASYNCcnt path, ISA 10.x / 15.18.3).
// lds_off = LDS byte address (low 32 bits of generic pointer), g = global addr.
__device__ __forceinline__ void async_copy16(unsigned lds_off, const void* g) {
  asm volatile("global_load_async_to_lds_b128 %0, %1, off"
               :: "v"(lds_off), "v"(g) : "memory");
}

// ---------------------------------------------------------------------------
// fp32 -> bf16 conversion
// ---------------------------------------------------------------------------
__global__ __launch_bounds__(256) void k_cvt(const float* __restrict__ src,
                                             u16* __restrict__ dst, int n) {
  int i = blockIdx.x * blockDim.x + threadIdx.x;
  int stride = gridDim.x * blockDim.x;
  for (; i < n; i += stride) dst[i] = f32_to_bf16(src[i]);
}

// ---------------------------------------------------------------------------
// C[M,N] = A[M,K] * B[N,K]^T   (NT GEMM, bf16 in, f32 out)
// WG = 256 threads = 8 waves (2M x 4N); WG tile 128x128.
// Stages of K=64 are DMA'd global->LDS with global_load_async_to_lds_b128,
// double-buffered (72 KB LDS) and overlapped with WMMA compute; fragments are
// re-read from LDS (ds_load_b128, 16B-padded rows for bank spread).
// ---------------------------------------------------------------------------
__global__ __launch_bounds__(256) void k_gemm_nt(const u16* __restrict__ A,
                                                 const u16* __restrict__ B,
                                                 float* __restrict__ C,
                                                 int M, int N, int K) {
  __shared__ __align__(16) u16 Asm[2][128 * LDSROW];
  __shared__ __align__(16) u16 Bsm[2][128 * LDSROW];

  const int tid  = threadIdx.x;
  const int lane = tid & 31;
  const int wv   = tid >> 5;
  const int hlf  = lane >> 4;
  const int r    = lane & 15;
  const int m_base = blockIdx.y * 128;
  const int n_base = blockIdx.x * 128;
  const int wm = (wv & 1) * 64;    // wave M offset within WG tile
  const int wn = (wv >> 1) * 32;   // wave N offset within WG tile

  // Copy-unit decomposition for one 128x64 bf16 tile (16 KB = 1024 x 16B):
  // unit u = p*256 + tid ; row = u>>3 ; seg = u&7 (16B segment within row).
  const int cp_row = tid >> 3;          // rows covered per pass: tid/8 + p*32
  const int cp_seg = tid & 7;
  const unsigned a_lds0 = (unsigned)(size_t)&Asm[0][0];
  const unsigned b_lds0 = (unsigned)(size_t)&Bsm[0][0];
  const unsigned stage_bytes = 128 * LDSROW * 2;

  v8f acc[4][2];
  #pragma unroll
  for (int mi = 0; mi < 4; ++mi)
    #pragma unroll
    for (int ni = 0; ni < 2; ++ni) acc[mi][ni] = v8f_zero();

  auto issue_stage = [&](int buf, int k0) {
    #pragma unroll
    for (int p = 0; p < 4; ++p) {
      const int row = cp_row + p * 32;
      async_copy16(a_lds0 + buf * stage_bytes + row * (LDSROW * 2) + cp_seg * 16,
                   A + (size_t)(m_base + row) * K + k0 + cp_seg * 8);
    }
    #pragma unroll
    for (int p = 0; p < 4; ++p) {
      const int row = cp_row + p * 32;
      async_copy16(b_lds0 + buf * stage_bytes + row * (LDSROW * 2) + cp_seg * 16,
                   B + (size_t)(n_base + row) * K + k0 + cp_seg * 8);
    }
  };

  issue_stage(0, 0);

  const int nst = K / KB_STAGE;
  for (int i = 0; i < nst; ++i) {
    const int buf = i & 1;
    if (i + 1 < nst) {
      issue_stage(buf ^ 1, (i + 1) * KB_STAGE);
      asm volatile("s_wait_asynccnt 0x8" ::: "memory");  // stage i landed (in-order)
    } else {
      asm volatile("s_wait_asynccnt 0x0" ::: "memory");
    }
    __syncthreads();                     // all waves' copies for stage i visible

    const u16* As = &Asm[buf][0];
    const u16* Bs = &Bsm[buf][0];
    #pragma unroll
    for (int kk = 0; kk < 2; ++kk) {
      v16bf af[4], bfm[2];
      #pragma unroll
      for (int mi = 0; mi < 4; ++mi)
        af[mi] = load_frag(As + (wm + mi * 16 + r) * LDSROW + kk * 32 + hlf * 8);
      #pragma unroll
      for (int ni = 0; ni < 2; ++ni)
        bfm[ni] = load_frag(Bs + (wn + ni * 16 + r) * LDSROW + kk * 32 + hlf * 8);
      #pragma unroll
      for (int mi = 0; mi < 4; ++mi)
        #pragma unroll
        for (int ni = 0; ni < 2; ++ni)
          acc[mi][ni] = __builtin_amdgcn_wmma_f32_16x16x32_bf16(
              false, af[mi], false, bfm[ni], (short)0, acc[mi][ni], false, false);
    }
    __syncthreads();                     // buffer reusable by stage i+2 copies
  }

  // C layout: lane holds column n_base+wn+ni*16+r, rows m_base+wm+mi*16+hlf*8+j
  #pragma unroll
  for (int mi = 0; mi < 4; ++mi)
    #pragma unroll
    for (int ni = 0; ni < 2; ++ni)
      #pragma unroll
      for (int j = 0; j < 8; ++j)
        C[(size_t)(m_base + wm + mi * 16 + hlf * 8 + j) * N
          + (n_base + wn + ni * 16 + r)] = acc[mi][ni][j];
}

// ---------------------------------------------------------------------------
// Fused causal depthwise conv(K=4) + SiLU, RMS-norm (q,k), RoPE (q,k).
// One block = one (t, h); 128 threads = one channel each.
// Emits Qb/Kb bf16 [H][T][HD], V transposed bf16 [H][HD][T].
// ---------------------------------------------------------------------------
__device__ __forceinline__ float conv_silu(const float* __restrict__ x,
                                           const float* __restrict__ w,
                                           int t, int c) {
  float acc = 0.f;
  #pragma unroll
  for (int j = 0; j < 4; ++j) {
    int tt = t - 3 + j;
    float xv = (tt >= 0) ? x[(size_t)tt * D_DIM + c] : 0.f;
    acc += xv * w[c * 4 + j];
  }
  return acc / (1.f + __expf(-acc));   // SiLU
}

__global__ __launch_bounds__(128) void k_post(
    const float* __restrict__ q32, const float* __restrict__ k32,
    const float* __restrict__ v32,
    const float* __restrict__ cwq, const float* __restrict__ cwk,
    const float* __restrict__ cwv,
    const float* __restrict__ qnw, const float* __restrict__ knw,
    const float* __restrict__ cosb, const float* __restrict__ sinb,
    u16* __restrict__ Qb, u16* __restrict__ Kb, u16* __restrict__ Vt) {
  const int t = blockIdx.x;
  const int h = blockIdx.y;
  const int d = threadIdx.x;
  const int c = h * HD_N + d;

  __shared__ float sred[HD_N];
  __shared__ float sval[HD_N];

  const float yq = conv_silu(q32, cwq, t, c);
  const float yk = conv_silu(k32, cwk, t, c);
  const float yv = conv_silu(v32, cwv, t, c);

  const float cs = (d < RD_N) ? cosb[(size_t)t * RD_N + d] : 0.f;
  const float sn = (d < RD_N) ? sinb[(size_t)t * RD_N + d] : 0.f;

  // ---- Q: rms-norm + rope ----
  sred[d] = yq * yq;
  __syncthreads();
  for (int st = HD_N / 2; st > 0; st >>= 1) {
    if (d < st) sred[d] += sred[d + st];
    __syncthreads();
  }
  float rq = rsqrtf(sred[0] * (1.f / HD_N) + EPS_F);
  float qn = yq * rq * qnw[d];
  __syncthreads();
  sval[d] = qn;
  __syncthreads();
  float qout;
  if (d < RD_N) {
    float rh = (d < RD_N / 2) ? -sval[d + RD_N / 2] : sval[d - RD_N / 2];
    qout = -(qn * cs + rh * sn);   // reference negates the rotated part
  } else {
    qout = qn;
  }
  Qb[((size_t)h * T_DIM + t) * HD_N + d] = f32_to_bf16(qout);
  __syncthreads();

  // ---- K: rms-norm + rope ----
  sred[d] = yk * yk;
  __syncthreads();
  for (int st = HD_N / 2; st > 0; st >>= 1) {
    if (d < st) sred[d] += sred[d + st];
    __syncthreads();
  }
  float rk = rsqrtf(sred[0] * (1.f / HD_N) + EPS_F);
  float kn = yk * rk * knw[d];
  __syncthreads();
  sval[d] = kn;
  __syncthreads();
  float kout;
  if (d < RD_N) {
    float rh = (d < RD_N / 2) ? -sval[d + RD_N / 2] : sval[d - RD_N / 2];
    kout = -(kn * cs + rh * sn);
  } else {
    kout = kn;
  }
  Kb[((size_t)h * T_DIM + t) * HD_N + d] = f32_to_bf16(kout);

  // ---- V: transposed layout for contiguous B-fragment loads in P@V ----
  Vt[(size_t)(h * HD_N + d) * T_DIM + t] = f32_to_bf16(yv);
}

// ---------------------------------------------------------------------------
// Flash attention. One wave = 16 query rows; KV streamed in 64-wide blocks.
// S = Q K^T via WMMA (NT), online softmax with row stats in registers
// (cross-lane reduce within the 16-lane half = the C-matrix column group),
// P re-shaped C-layout -> A-layout through a per-wave LDS tile, O += P V
// via WMMA against the [H][HD][T] transposed V.
// ---------------------------------------------------------------------------
__global__ __launch_bounds__(256) void k_attn(const u16* __restrict__ Qb,
                                              const u16* __restrict__ Kb,
                                              const u16* __restrict__ Vt,
                                              u16* __restrict__ Ob) {
  __shared__ __align__(16) u16 Pbuf[8][16 * 64];   // per-wave 16x64 bf16 tile
  const int lane = threadIdx.x & 31;
  const int wv   = threadIdx.x >> 5;
  const int hlf  = lane >> 4;
  const int r    = lane & 15;
  const int h    = blockIdx.y;
  const int q0   = (blockIdx.x * 8 + wv) * 16;

  const u16* Qh = Qb + (size_t)h * T_DIM * HD_N;
  const u16* Kh = Kb + (size_t)h * T_DIM * HD_N;
  const u16* Vh = Vt + (size_t)h * HD_N * T_DIM;

  // Cache Q fragments for all 4 K-steps of HD=128
  v16bf qf[4];
  #pragma unroll
  for (int kk = 0; kk < 4; ++kk)
    qf[kk] = load_frag(Qh + (size_t)(q0 + r) * HD_N + kk * 32 + hlf * 8);

  v8f o[8];
  #pragma unroll
  for (int di = 0; di < 8; ++di) o[di] = v8f_zero();
  float mrow[8], lrow[8];
  #pragma unroll
  for (int j = 0; j < 8; ++j) { mrow[j] = -1e30f; lrow[j] = 0.f; }

  const float scale = 0.088388347648318447f;   // 1/sqrt(128)
  const int kend = q0 + 16;                    // causal bound (exclusive)

  for (int kb = 0; kb < kend; kb += 64) {
    // ---- S tiles: 4 x (16x16), K-dim = HD in 4 steps ----
    v8f s[4];
    #pragma unroll
    for (int tn = 0; tn < 4; ++tn) {
      v8f st = v8f_zero();
      #pragma unroll
      for (int kk = 0; kk < 4; ++kk) {
        v16bf kf = load_frag(Kh + (size_t)(kb + tn * 16 + r) * HD_N + kk * 32 + hlf * 8);
        st = __builtin_amdgcn_wmma_f32_16x16x32_bf16(
            false, qf[kk], false, kf, (short)0, st, false, false);
      }
      s[tn] = st;
    }
    // ---- scale + causal mask (row = q0+hlf*8+j, col = kb+tn*16+r) ----
    #pragma unroll
    for (int tn = 0; tn < 4; ++tn) {
      const int ki = kb + tn * 16 + r;
      #pragma unroll
      for (int j = 0; j < 8; ++j) {
        const int qi = q0 + hlf * 8 + j;
        float v = s[tn][j] * scale;
        s[tn][j] = (ki > qi) ? -1e30f : v;
      }
    }
    // ---- online softmax: row max / sum across the 16-lane half ----
    float alpha[8];
    #pragma unroll
    for (int j = 0; j < 8; ++j) {
      float mx = fmaxf(fmaxf(s[0][j], s[1][j]), fmaxf(s[2][j], s[3][j]));
      mx = fmaxf(mx, __shfl_xor(mx, 1));
      mx = fmaxf(mx, __shfl_xor(mx, 2));
      mx = fmaxf(mx, __shfl_xor(mx, 4));
      mx = fmaxf(mx, __shfl_xor(mx, 8));
      const float mnew = fmaxf(mrow[j], mx);
      alpha[j] = __expf(mrow[j] - mnew);
      mrow[j] = mnew;
    }
    #pragma unroll
    for (int tn = 0; tn < 4; ++tn)
      #pragma unroll
      for (int j = 0; j < 8; ++j)
        s[tn][j] = __expf(s[tn][j] - mrow[j]);
    #pragma unroll
    for (int j = 0; j < 8; ++j) {
      float sm = s[0][j] + s[1][j] + s[2][j] + s[3][j];
      sm += __shfl_xor(sm, 1);
      sm += __shfl_xor(sm, 2);
      sm += __shfl_xor(sm, 4);
      sm += __shfl_xor(sm, 8);
      lrow[j] = lrow[j] * alpha[j] + sm;
    }
    // ---- stash P (bf16) to LDS in row-major [16][64]; rescale O ----
    #pragma unroll
    for (int tn = 0; tn < 4; ++tn)
      #pragma unroll
      for (int j = 0; j < 8; ++j)
        Pbuf[wv][(hlf * 8 + j) * 64 + tn * 16 + r] = f32_to_bf16(s[tn][j]);
    #pragma unroll
    for (int di = 0; di < 8; ++di)
      #pragma unroll
      for (int j = 0; j < 8; ++j)
        o[di][j] *= alpha[j];
    __asm__ volatile("s_wait_dscnt 0" ::: "memory");
    // ---- O += P @ V : A-frags from LDS, B-frags contiguous from Vt ----
    #pragma unroll
    for (int kk = 0; kk < 2; ++kk) {
      v16bf pf = load_frag(&Pbuf[wv][r * 64 + kk * 32 + hlf * 8]);
      #pragma unroll
      for (int di = 0; di < 8; ++di) {
        v16bf vf = load_frag(Vh + (size_t)(di * 16 + r) * T_DIM + kb + kk * 32 + hlf * 8);
        o[di] = __builtin_amdgcn_wmma_f32_16x16x32_bf16(
            false, pf, false, vf, (short)0, o[di], false, false);
      }
    }
  }

  float inv[8];
  #pragma unroll
  for (int j = 0; j < 8; ++j) inv[j] = 1.f / lrow[j];
  #pragma unroll
  for (int di = 0; di < 8; ++di)
    #pragma unroll
    for (int j = 0; j < 8; ++j)
      Ob[(size_t)(q0 + hlf * 8 + j) * D_DIM + h * HD_N + di * 16 + r]
          = f32_to_bf16(o[di][j] * inv[j]);
}

// ---------------------------------------------------------------------------
// Host orchestration
// ---------------------------------------------------------------------------
extern "C" void kernel_launch(void* const* d_in, const int* in_sizes, int n_in,
                              void* d_out, int out_size, void* d_ws, size_t ws_size,
                              hipStream_t stream) {
  (void)in_sizes; (void)n_in; (void)out_size; (void)ws_size;
  const float* hidden = (const float*)d_in[0];
  const float* cosb   = (const float*)d_in[1];
  const float* sinb   = (const float*)d_in[2];
  const float* Wq     = (const float*)d_in[3];
  const float* Wk     = (const float*)d_in[4];
  const float* Wv     = (const float*)d_in[5];
  const float* Wo     = (const float*)d_in[6];
  const float* cwq    = (const float*)d_in[7];
  const float* cwk    = (const float*)d_in[8];
  const float* cwv    = (const float*)d_in[9];
  const float* qnw    = (const float*)d_in[10];
  const float* knw    = (const float*)d_in[11];

  const size_t TD = (size_t)T_DIM * D_DIM;
  const size_t BF = TD * sizeof(u16);    // bf16 matrix bytes (8 MiB)
  const size_t F4 = TD * sizeof(float);  // f32 matrix bytes (16 MiB)
  char* ws = (char*)d_ws;
  u16*   WqB = (u16*)(ws + 0 * BF);
  u16*   WkB = (u16*)(ws + 1 * BF);
  u16*   WvB = (u16*)(ws + 2 * BF);
  u16*   WoB = (u16*)(ws + 3 * BF);
  u16*   Xb  = (u16*)(ws + 4 * BF);
  float* q32 = (float*)(ws + 5 * BF);
  float* k32 = (float*)(ws + 5 * BF + 1 * F4);
  float* v32 = (float*)(ws + 5 * BF + 2 * F4);
  u16*   Qb  = (u16*)(ws + 5 * BF + 3 * F4);
  u16*   Kb  = (u16*)(ws + 6 * BF + 3 * F4);
  u16*   Vt  = (u16*)(ws + 7 * BF + 3 * F4);
  u16*   Ob  = Xb;   // Xb dead after QKV GEMMs -> reuse for attention output

  // 1) bf16 conversions
  k_cvt<<<dim3(1024), dim3(256), 0, stream>>>(hidden, Xb, (int)TD);
  k_cvt<<<dim3(1024), dim3(256), 0, stream>>>(Wq, WqB, (int)TD);
  k_cvt<<<dim3(1024), dim3(256), 0, stream>>>(Wk, WkB, (int)TD);
  k_cvt<<<dim3(1024), dim3(256), 0, stream>>>(Wv, WvB, (int)TD);
  k_cvt<<<dim3(1024), dim3(256), 0, stream>>>(Wo, WoB, (int)TD);

  // 2) QKV projections: [T,D] = [T,D] @ W^T
  dim3 ggrd(D_DIM / 128, T_DIM / 128);
  k_gemm_nt<<<ggrd, dim3(256), 0, stream>>>(Xb, WqB, q32, T_DIM, D_DIM, D_DIM);
  k_gemm_nt<<<ggrd, dim3(256), 0, stream>>>(Xb, WkB, k32, T_DIM, D_DIM, D_DIM);
  k_gemm_nt<<<ggrd, dim3(256), 0, stream>>>(Xb, WvB, v32, T_DIM, D_DIM, D_DIM);

  // 3) conv + SiLU + RMS-norm + RoPE, layout change
  k_post<<<dim3(T_DIM, H_N), dim3(HD_N), 0, stream>>>(
      q32, k32, v32, cwq, cwk, cwv, qnw, knw, cosb, sinb, Qb, Kb, Vt);

  // 4) causal flash attention
  k_attn<<<dim3(T_DIM / 128, H_N), dim3(256), 0, stream>>>(Qb, Kb, Vt, Ob);

  // 5) output projection -> f32 d_out
  k_gemm_nt<<<ggrd, dim3(256), 0, stream>>>(Ob, WoB, (float*)d_out, T_DIM, D_DIM, D_DIM);
}
